// DLRM_72859825209705
// MI455X (gfx1250) — compile-verified, matrix operand order checked by reading, not run on hardware
//
#include <hip/hip_runtime.h>
#include <hip/hip_bf16.h>

// ---------------- problem constants (match reference) ----------------
#define T_TABLES   26
#define D_EMB      64
#define B_BATCH    4096
#define L_LOOK     32
#define SLOTS      27            // 1 (dense x) + 26 pooled tables
#define TCAT_SLOTS 32            // padded to 32 rows (27..31 zeroed)
#define TCAT_STRIDE (TCAT_SLOTS * D_EMB)   // 2048 bf16 / sample
#define R_COLS     415           // 64 + 27*26/2
#define R_STRIDE   416           // padded to 13*32 for clean K

typedef __attribute__((ext_vector_type(16))) __bf16 v16bf;
typedef __attribute__((ext_vector_type(8)))  __bf16 v8bf;
typedef __attribute__((ext_vector_type(8)))  float  v8f;

// Fragment load: wave-striped 16x32 bf16 tile, lane = row (A) / column (B).
// Elements 0..7 = K[koff..koff+7], 8..15 = K[koff+16..koff+23], koff = lane>=16 ? 8 : 0.
// Two 16-byte global_load_b128 per lane, no guards, no converts.
__device__ __forceinline__ v16bf ldfrag(const __bf16* __restrict__ p, int lane) {
  const __bf16* q = p + ((lane & 16) ? 8 : 0);
  v8bf lo = *(const v8bf*)q;
  v8bf hi = *(const v8bf*)(q + 16);
  return __builtin_shufflevector(lo, hi, 0, 1, 2, 3, 4, 5, 6, 7,
                                 8, 9, 10, 11, 12, 13, 14, 15);
}

#define WMMA_BF16(A, B, C) \
  __builtin_amdgcn_wmma_f32_16x16x32_bf16(false, (A), false, (B), (short)0, (C), false, false)

// ---------------- Y = relu(X @ W^T + bias) ----------------
// X: (M,Kpad) bf16 row stride ldx; W: (Npad,Kpad) bf16 packed+padded; bias f32.
// Each wave: one 32x32 superblock; 8 waves / block. M%32==0, Npad%32==0, Kpad%32==0.
template <bool OUT_F32>
__global__ void gemm_bf16_wmma(const __bf16* __restrict__ X,
                               const __bf16* __restrict__ W,
                               const float* __restrict__ bias,
                               void* __restrict__ Yv,
                               int M, int Nlog, int Npad, int Kpad,
                               int ldx, int ldy) {
  const int wave = threadIdx.x >> 5;
  const int lane = threadIdx.x & 31;
  const int nsb = Npad >> 5;
  const int sb = blockIdx.x * 8 + wave;
  if (sb >= (M >> 5) * nsb) return;             // wave-uniform
  const int msb = sb / nsb;
  const int nsi = sb - msb * nsb;
  const int m0 = msb << 5, n0 = nsi << 5;
  const int l15 = lane & 15;

  const __bf16* a0p = X + (size_t)(m0 + l15) * ldx;
  const __bf16* a1p = X + (size_t)(m0 + 16 + l15) * ldx;
  const __bf16* b0p = W + (size_t)(n0 + l15) * Kpad;
  const __bf16* b1p = W + (size_t)(n0 + 16 + l15) * Kpad;

  v8f c00 = {}, c01 = {}, c10 = {}, c11 = {};
  for (int kb = 0; kb < Kpad; kb += 32) {
    v16bf a0 = ldfrag(a0p + kb, lane);
    v16bf a1 = ldfrag(a1p + kb, lane);
    v16bf b0 = ldfrag(b0p + kb, lane);
    v16bf b1 = ldfrag(b1p + kb, lane);
    c00 = WMMA_BF16(a0, b0, c00);
    c01 = WMMA_BF16(a0, b1, c01);
    c10 = WMMA_BF16(a1, b0, c10);
    c11 = WMMA_BF16(a1, b1, c11);
  }

  const int mb = (lane & 16) ? 8 : 0;
#pragma unroll
  for (int t = 0; t < 4; ++t) {
    const v8f& c = (t == 0) ? c00 : (t == 1) ? c01 : (t == 2) ? c10 : c11;
    const int col = n0 + ((t & 1) << 4) + l15;
    const int mrow = m0 + ((t >> 1) << 4) + mb;
    if (col < Nlog) {
      const float bv = bias[col];
#pragma unroll
      for (int i = 0; i < 8; ++i) {
        float v = c[i] + bv;
        v = v > 0.0f ? v : 0.0f;
        if constexpr (OUT_F32)
          ((float*)Yv)[(size_t)(mrow + i) * ldy + col] = v;
        else
          ((__bf16*)Yv)[(size_t)(mrow + i) * ldy + col] = (__bf16)v;
      }
    }
  }
}

// ---------------- f32 -> bf16 pack with zero padding ----------------
__global__ void pack_pad_bf16(const float* __restrict__ in, __bf16* __restrict__ out,
                              int rIn, int cIn, int cOut, int total) {
  int idx = blockIdx.x * 256 + threadIdx.x;
  if (idx >= total) return;
  int r = idx / cOut, c = idx - r * cOut;
  float v = (r < rIn && c < cIn) ? in[(size_t)r * cIn + c] : 0.0f;
  out[idx] = (__bf16)v;
}

// ---------------- zero Tcat pad slots 27..31 ----------------
__global__ void zero_tcat_pad(__bf16* __restrict__ Tcat) {
  int idx = blockIdx.x * 256 + threadIdx.x;       // B * 5 * 64 total, exact
  int b = idx / (5 * D_EMB);
  int rem = idx - b * (5 * D_EMB);
  Tcat[(size_t)b * TCAT_STRIDE + (SLOTS + rem / D_EMB) * D_EMB + (rem & 63)] =
      (__bf16)0.0f;
}

// ---------------- embedding gather + pool (f32 accum, bf16 store) ------
__global__ void embed_pool_kernel(const float* __restrict__ tables,
                                  const int* __restrict__ indices,
                                  __bf16* __restrict__ Tcat, int Nrows) {
  const int d = threadIdx.x;
  const int b = blockIdx.x;
  const int t = blockIdx.y;
  __shared__ int sidx[L_LOOK];
  if (threadIdx.x < L_LOOK)
    sidx[threadIdx.x] = indices[((size_t)t * B_BATCH + b) * L_LOOK + threadIdx.x];
  __syncthreads();
  const float* tab = tables + (size_t)t * Nrows * D_EMB;
  float acc = 0.0f;
#pragma unroll 8
  for (int l = 0; l < L_LOOK; ++l)
    acc += tab[(size_t)sidx[l] * D_EMB + d];
  Tcat[(size_t)b * TCAT_STRIDE + (1 + t) * D_EMB + d] = (__bf16)acc;
}

// ---------------- interaction: strict lower triangle of Tcat Tcat^T ------
__device__ __forceinline__ void store_tril(__bf16* __restrict__ Rz, const v8f& c,
                                           int itile, int jtile, int lane) {
  const int j = jtile * 16 + (lane & 15);
  const int ibase = itile * 16 + ((lane & 16) ? 8 : 0);
#pragma unroll
  for (int k = 0; k < 8; ++k) {
    int i = ibase + k;
    if (i < SLOTS && j < i)
      Rz[(i * (i - 1)) / 2 + j] = (__bf16)c[k];
  }
}

// one wave per sample; Z tiles (0,0),(1,0),(1,1); tile (0,1) has no i>j entries.
__global__ void interact_kernel(const __bf16* __restrict__ Tcat,
                                __bf16* __restrict__ R) {
  const int wave = threadIdx.x >> 5;
  const int lane = threadIdx.x & 31;
  const int b = blockIdx.x * 8 + wave;
  const __bf16* Tb = Tcat + (size_t)b * TCAT_STRIDE;
  __bf16* Rb = R + (size_t)b * R_STRIDE;

  Rb[lane]      = Tb[lane];            // R[0:64] = x (slot 0), already bf16
  Rb[32 + lane] = Tb[32 + lane];
  if (lane == 0) Rb[R_COLS] = (__bf16)0.0f;   // K padding column

  const int r = lane & 15;
  const __bf16* row0 = Tb + r * D_EMB;              // rows 0..15
  const __bf16* row1 = Tb + (16 + r) * D_EMB;       // rows 16..31 (27..31 zeroed)
  v8f c00 = {}, c10 = {}, c11 = {};
#pragma unroll
  for (int kb = 0; kb < D_EMB; kb += 32) {
    v16bf f0 = ldfrag(row0 + kb, lane);   // serves as both A and B fragment
    v16bf f1 = ldfrag(row1 + kb, lane);
    c00 = WMMA_BF16(f0, f0, c00);
    c10 = WMMA_BF16(f1, f0, c10);
    c11 = WMMA_BF16(f1, f1, c11);
  }
  __bf16* Rz = Rb + D_EMB;
  store_tril(Rz, c00, 0, 0, lane);
  store_tril(Rz, c10, 1, 0, lane);
  store_tril(Rz, c11, 1, 1, lane);
}

// ---------------- host-side orchestration ----------------
extern "C" void kernel_launch(void* const* d_in, const int* in_sizes, int n_in,
                              void* d_out, int out_size, void* d_ws, size_t ws_size,
                              hipStream_t stream) {
  const float* dense_x = (const float*)d_in[0];   // (4096, 13)
  const int*   indices = (const int*)  d_in[1];   // (26, 4096, 32)
  const float* tables  = (const float*)d_in[2];   // (26, N, 64)
  const float* bW0 = (const float*)d_in[3];  const float* bb0 = (const float*)d_in[4];
  const float* bW1 = (const float*)d_in[5];  const float* bb1 = (const float*)d_in[6];
  const float* bW2 = (const float*)d_in[7];  const float* bb2 = (const float*)d_in[8];
  const float* tW0 = (const float*)d_in[9];  const float* tb0 = (const float*)d_in[10];
  const float* tW1 = (const float*)d_in[11]; const float* tb1 = (const float*)d_in[12];
  const float* tW2 = (const float*)d_in[13]; const float* tb2 = (const float*)d_in[14];
  float* out = (float*)d_out;                     // (4096, 1)

  const int Nrows = in_sizes[2] / (T_TABLES * D_EMB);   // 100000
  const int M = B_BATCH;

  // ---- workspace layout (all bf16 elements, every offset 16B-aligned) ----
  __bf16* p = (__bf16*)d_ws;
  __bf16* Xp   = p; p += (size_t)M * 32;            // dense_x padded 13->32
  __bf16* W0p  = p; p += (size_t)512 * 32;          // bot_W0  512x13 -> 512x32
  __bf16* W1p  = p; p += (size_t)256 * 512;         // bot_W1
  __bf16* W2p  = p; p += (size_t)64 * 256;          // bot_W2
  __bf16* TW0p = p; p += (size_t)512 * R_STRIDE;    // top_W0  512x415 -> 512x416
  __bf16* TW1p = p; p += (size_t)256 * 512;         // top_W1
  __bf16* TW2p = p; p += (size_t)32 * 256;          // top_W2  1x256 -> 32x256
  __bf16* act512 = p; p += (size_t)M * 512;
  __bf16* act256 = p; p += (size_t)M * 256;
  __bf16* Tcat   = p; p += (size_t)M * TCAT_STRIDE;
  __bf16* Rbuf   = p; p += (size_t)M * R_STRIDE;

  auto packb = [](int total) { return (total + 255) / 256; };
  // ---- pack weights + dense input to padded bf16 ----
  pack_pad_bf16<<<packb(M * 32), 256, 0, stream>>>(dense_x, Xp, M, 13, 32, M * 32);
  pack_pad_bf16<<<packb(512 * 32), 256, 0, stream>>>(bW0, W0p, 512, 13, 32, 512 * 32);
  pack_pad_bf16<<<packb(256 * 512), 256, 0, stream>>>(bW1, W1p, 256, 512, 512, 256 * 512);
  pack_pad_bf16<<<packb(64 * 256), 256, 0, stream>>>(bW2, W2p, 64, 256, 256, 64 * 256);
  pack_pad_bf16<<<packb(512 * R_STRIDE), 256, 0, stream>>>(tW0, TW0p, 512, R_COLS,
                                                           R_STRIDE, 512 * R_STRIDE);
  pack_pad_bf16<<<packb(256 * 512), 256, 0, stream>>>(tW1, TW1p, 256, 512, 512, 256 * 512);
  pack_pad_bf16<<<packb(32 * 256), 256, 0, stream>>>(tW2, TW2p, 1, 256, 256, 32 * 256);

  auto gblocks = [M](int Npad) { return ((M >> 5) * (Npad >> 5) + 7) / 8; };

  // ---- bottom MLP: 13 -> 512 -> 256 -> 64 (layer 2 writes Tcat slot 0) ----
  gemm_bf16_wmma<false><<<gblocks(512), 256, 0, stream>>>(
      Xp, W0p, bb0, act512, M, 512, 512, 32, 32, 512);
  gemm_bf16_wmma<false><<<gblocks(256), 256, 0, stream>>>(
      act512, W1p, bb1, act256, M, 256, 256, 512, 512, 256);
  gemm_bf16_wmma<false><<<gblocks(64), 256, 0, stream>>>(
      act256, W2p, bb2, Tcat, M, 64, 64, 256, 256, TCAT_STRIDE);

  // ---- zero Tcat pad slots + embedding gather/pool into slots 1..26 ----
  zero_tcat_pad<<<(B_BATCH * 5 * D_EMB) / 256, 256, 0, stream>>>(Tcat);
  dim3 eg(B_BATCH, T_TABLES);
  embed_pool_kernel<<<eg, D_EMB, 0, stream>>>(tables, indices, Tcat, Nrows);

  // ---- pairwise interaction + concat -> R (bf16, stride 416) ----
  interact_kernel<<<B_BATCH / 8, 256, 0, stream>>>(Tcat, Rbuf);

  // ---- top MLP: 415 -> 512 -> 256 -> 1 (final layer f32 to d_out) ----
  gemm_bf16_wmma<false><<<gblocks(512), 256, 0, stream>>>(
      Rbuf, TW0p, tb0, act512, M, 512, 512, R_STRIDE, R_STRIDE, 512);
  gemm_bf16_wmma<false><<<gblocks(256), 256, 0, stream>>>(
      act512, TW1p, tb1, act256, M, 256, 256, 512, 512, 256);
  gemm_bf16_wmma<true><<<gblocks(32), 256, 0, stream>>>(
      act256, TW2p, tb2, out, M, 1, 32, 256, 256, 1);
}
